// SKAttention_53833120088360
// MI455X (gfx1250) — matrix operand order repaired and checked by reading, not compiled, not run on hardware
//
#include <hip/hip_runtime.h>
#include <hip/hip_bf16.h>

#define B_   4
#define C_   768
#define H_   56
#define W_   56
#define P_   3136      // H_*W_
#define C8_  96
#define D_   48
#define BN_EPS 1e-5f

typedef __attribute__((ext_vector_type(16))) _Float16 v16h;
typedef __attribute__((ext_vector_type(8)))  _Float16 v8h;
typedef __attribute__((ext_vector_type(8)))  float    v8f;

// TDM descriptor group types (probe-confirmed signatures)
typedef __attribute__((ext_vector_type(4))) unsigned int tdm_g0_t;
typedef __attribute__((ext_vector_type(8))) int          tdm_g1_t;
typedef __attribute__((ext_vector_type(4))) int          tdm_g2_t;

#if __has_include(<hip/amd_detail/amd_gfx1250_TDM.h>)
#define TDM_SIX_ARGS 1
#else
#define TDM_SIX_ARGS 0
#endif

// Issue a 2D tensor tile DMA (global -> LDS). Groups 2/3 zero-filled => 2D.
__device__ __forceinline__ void tdm_load_2d(tdm_g0_t g0, tdm_g1_t g1) {
  tdm_g2_t z4 = {};
#if TDM_SIX_ARGS
  tdm_g1_t z8 = {};
  __builtin_amdgcn_tensor_load_to_lds(g0, g1, z4, z4, z8, 0);
#else
  __builtin_amdgcn_tensor_load_to_lds(g0, g1, z4, z4, 0);
#endif
}

// ---------------------------------------------------------------------------
// A-fragment loader: 16x32 f16 A-matrix, row-major source with stride C.
// Layout: lanes 0-15 -> M=lane, K 0..15; lanes 16-31 ->
// M=lane-16, K 16..31.  kb = (lane<16)?0:16.
// ---------------------------------------------------------------------------
__device__ __forceinline__ v16h load_a_frag(const _Float16* row, int k0, int kb) {
  v8h lo = *(const v8h*)(row + k0 + kb);
  v8h hi = *(const v8h*)(row + k0 + 8 + kb);
  return __builtin_shufflevector(lo, hi, 0, 1, 2, 3, 4, 5, 6, 7,
                                 8, 9, 10, 11, 12, 13, 14, 15);
}

// ---------------------------------------------------------------------------
// Convert (B,C,P) f32 -> (B,P,C) f16  (transpose for WMMA B-fragments)
// ---------------------------------------------------------------------------
__global__ __launch_bounds__(256) void cvt_xt_kernel(
    const float* __restrict__ x, _Float16* __restrict__ xt) {
  size_t idx = (size_t)blockIdx.x * 256 + threadIdx.x;
  if (idx >= (size_t)B_ * P_ * C_) return;
  int c = (int)(idx % C_);
  size_t t2 = idx / C_;
  int p = (int)(t2 % P_);
  int b = (int)(t2 / P_);
  xt[idx] = (_Float16)x[((size_t)(b * C_ + c)) * P_ + p];
}

__global__ __launch_bounds__(256) void cvt_f16_kernel(
    const float* __restrict__ in, _Float16* __restrict__ out, int n) {
  int i = blockIdx.x * 256 + threadIdx.x;
  if (i < n) out[i] = (_Float16)in[i];
}

// Pack conv weight (O,C,k,k) f32 -> tap-major (k*k, O, C) f16
__global__ __launch_bounds__(256) void pack_conv_w_kernel(
    const float* __restrict__ w, _Float16* __restrict__ wt, int ks) {
  size_t n = (size_t)C_ * C_ * ks * ks;
  size_t idx = (size_t)blockIdx.x * 256 + threadIdx.x;
  if (idx >= n) return;
  int c = (int)(idx % C_);
  size_t t2 = idx / C_;
  int o = (int)(t2 % C_);
  int tap = (int)(t2 / C_);
  int dy = tap / ks, dx = tap % ks;
  wt[idx] = (_Float16)w[(((size_t)o * C_ + c) * ks + dy) * ks + dx];
}

// ---------------------------------------------------------------------------
// WMMA GEMM: out[b,o,p] = bias[o] + sum_c Wf[o,c] * Xt[b,p,c]
// Wave computes a 16x64 tile; blockDim = 32*NW waves tile M.
// grid.x = B * (O/(16*NW)) * (P/64)
// ---------------------------------------------------------------------------
__global__ __launch_bounds__(256) void gemm_bias_wmma(
    const _Float16* __restrict__ Wf, const float* __restrict__ bias,
    const _Float16* __restrict__ Xt, float* __restrict__ out, int O) {
  const int lane = threadIdx.x & 31;
  const int wid  = threadIdx.x >> 5;
  const int NW   = blockDim.x >> 5;
  const int NB   = P_ / 64;
  int blk = blockIdx.x;
  int nt = blk % NB; blk /= NB;
  int MB = O / (16 * NW);
  int mb = blk % MB;
  int b  = blk / MB;
  const int m0 = (mb * NW + wid) * 16;
  const int n0 = nt * 64;
  const int ln15 = lane & 15;
  const int kb   = (lane < 16) ? 0 : 16;
  const int nb16 = (lane < 16) ? 0 : 16;
  const _Float16* __restrict__ arow = Wf + (size_t)(m0 + ln15) * C_;
  const _Float16* __restrict__ Xb = Xt + (size_t)b * P_ * C_;
  const _Float16* bp0 = Xb + (size_t)(n0 + 0 * 16 + ln15) * C_ + nb16;
  const _Float16* bp1 = Xb + (size_t)(n0 + 1 * 16 + ln15) * C_ + nb16;
  const _Float16* bp2 = Xb + (size_t)(n0 + 2 * 16 + ln15) * C_ + nb16;
  const _Float16* bp3 = Xb + (size_t)(n0 + 3 * 16 + ln15) * C_ + nb16;
  v8f a0 = {}, a1 = {}, a2 = {}, a3 = {};
  for (int k0 = 0; k0 < C_; k0 += 32) {
    __builtin_prefetch(arow + k0 + 64, 0, 1);
    v16h a = load_a_frag(arow, k0, kb);
    v16h b0 = *(const v16h*)(bp0 + k0);
    v16h b1 = *(const v16h*)(bp1 + k0);
    v16h b2 = *(const v16h*)(bp2 + k0);
    v16h b3 = *(const v16h*)(bp3 + k0);
    a0 = __builtin_amdgcn_wmma_f32_16x16x32_f16(false, a, false, b0, (short)0, a0, false, false);
    a1 = __builtin_amdgcn_wmma_f32_16x16x32_f16(false, a, false, b1, (short)0, a1, false, false);
    a2 = __builtin_amdgcn_wmma_f32_16x16x32_f16(false, a, false, b2, (short)0, a2, false, false);
    a3 = __builtin_amdgcn_wmma_f32_16x16x32_f16(false, a, false, b3, (short)0, a3, false, false);
  }
  const int mmb = m0 + ((lane < 16) ? 0 : 8);
#pragma unroll
  for (int r = 0; r < 8; ++r) {
    int m = mmb + r;
    float bv = bias[m];
    size_t base = ((size_t)b * O + m) * P_;
    out[base + n0 + 0 * 16 + ln15] = a0[r] + bv;
    out[base + n0 + 1 * 16 + ln15] = a1[r] + bv;
    out[base + n0 + 2 * 16 + ln15] = a2[r] + bv;
    out[base + n0 + 3 * 16 + ln15] = a3[r] + bv;
  }
}

// ---------------------------------------------------------------------------
// Implicit-GEMM conv (SAME) + bias + BN + ReLU, with TDM-staged weights.
// Wt: (taps, O=C_, C_) f16; Xt: (B,P,C) f16; out feats: (B,C,P) f32
// Block = 128 threads (4 waves) covering 64 output channels x 64 pixels.
// Per tap: wave 0 DMAs the 64x768 f16 weight tile into LDS via the TDM
// (tensor_load_to_lds), waits TENSORcnt, then all waves consume A-fragments
// from LDS while streaming B-fragments from global.
// Dynamic LDS: 64*768*2 = 96 KB.
// ---------------------------------------------------------------------------
__global__ __launch_bounds__(128) void conv_bn_relu_wmma(
    const _Float16* __restrict__ Wt, const float* __restrict__ cb,
    const float* __restrict__ gg, const float* __restrict__ bb,
    const float* __restrict__ mm, const float* __restrict__ vv,
    const _Float16* __restrict__ Xt, float* __restrict__ out, int ks) {
  extern __shared__ char smem_raw[];
  _Float16* wsm = (_Float16*)smem_raw;      // 64 rows x 768 halves
  const int lane = threadIdx.x & 31;
  const int wid  = threadIdx.x >> 5;        // 0..3
  const int NB   = P_ / 64;                 // 49
  int blk = blockIdx.x;
  int nt = blk % NB; blk /= NB;
  const int MB = C_ / 64;                   // 12
  int mb = blk % MB;
  int b  = blk / MB;
  const int m0blk = mb * 64;
  const int m0 = m0blk + wid * 16;
  const int n0 = nt * 64;
  const int ln15 = lane & 15;
  const int kb   = (lane < 16) ? 0 : 16;
  const int nb16 = (lane < 16) ? 0 : 16;
  const _Float16* __restrict__ Xb = Xt + (size_t)b * P_ * C_;
  int ph[4], pw[4];
#pragma unroll
  for (int t = 0; t < 4; ++t) {
    int p = n0 + t * 16 + ln15;
    ph[t] = p / W_;
    pw[t] = p % W_;
  }
  const int rad = ks >> 1;
  const int ntaps = ks * ks;
  const unsigned lds_base = (unsigned)(size_t)wsm;   // LDS byte offset
  v8f acc[4] = {};
  for (int tap = 0; tap < ntaps; ++tap) {
    int dy = tap / ks - rad, dx = tap % ks - rad;
    size_t boff[4];
    bool ok[4];
#pragma unroll
    for (int t = 0; t < 4; ++t) {
      int hh = ph[t] + dy, ww = pw[t] + dx;
      ok[t] = (hh >= 0) && (hh < H_) && (ww >= 0) && (ww < W_);
      boff[t] = ok[t] ? ((size_t)(hh * W_ + ww) * C_ + nb16) : 0;
    }
    __syncthreads();   // prior tap's LDS reads complete before overwrite
    if (wid == 0) {
      unsigned long long ga =
          (unsigned long long)(size_t)(Wt + ((size_t)tap * C_ + m0blk) * C_);
      // D# group 0: count=1 | lds_addr | global_addr[56:0] | type=2
      tdm_g0_t g0 = { 1u,
                      lds_base,
                      (unsigned)(ga & 0xffffffffull),
                      (unsigned)((ga >> 32) & 0x1ffffffull) | 0x80000000u };
      // D# group 1: data_size=2B; tensor 768x64; tile 768x64; stride0=768
      tdm_g1_t g1 = { (int)(1u << 16),                       // data_size=1 (2B)
                      (int)(((unsigned)C_ & 0xffffu) << 16), // tensor_dim0 lo
                      (int)((64u & 0xffffu) << 16),          // dim0 hi=0 | dim1 lo
                      (int)(((unsigned)C_) << 16),           // dim1 hi=0 | tile_dim0
                      (int)64,                               // tile_dim1 | tile_dim2=0
                      (int)C_,                               // stride0[31:0]
                      0, 0 };
      tdm_load_2d(g0, g1);
      __builtin_amdgcn_s_wait_tensorcnt(0);
    }
    __syncthreads();   // weight tile visible to all waves
    const _Float16* arow = wsm + (size_t)(wid * 16 + ln15) * C_;
    for (int k0 = 0; k0 < C_; k0 += 32) {
      v16h a = load_a_frag(arow, k0, kb);
#pragma unroll
      for (int t = 0; t < 4; ++t) {
        v16h bf = {};
        if (ok[t]) bf = *(const v16h*)(Xb + boff[t] + k0);
        acc[t] = __builtin_amdgcn_wmma_f32_16x16x32_f16(false, a, false, bf,
                                                        (short)0, acc[t], false, false);
      }
    }
  }
  const int mmb = m0 + ((lane < 16) ? 0 : 8);
#pragma unroll
  for (int r = 0; r < 8; ++r) {
    int m = mmb + r;
    float inv  = gg[m] * rsqrtf(vv[m] + BN_EPS);
    float bias = cb[m];
    float mean = mm[m];
    float beta = bb[m];
    size_t base = ((size_t)b * C_ + m) * P_;
#pragma unroll
    for (int t = 0; t < 4; ++t) {
      float y = acc[t][r] + bias;
      y = (y - mean) * inv + beta;
      out[base + n0 + t * 16 + ln15] = fmaxf(y, 0.0f);
    }
  }
}

// ---------------------------------------------------------------------------
// Criss-cross energies + softmax.  One block per (b,i,w) row; 112 entries.
// ---------------------------------------------------------------------------
__global__ __launch_bounds__(128) void cc_energy_softmax(
    const float* __restrict__ q, const float* __restrict__ kk_,
    float* __restrict__ attn) {
  int row = blockIdx.x;                 // ((b*H)+i)*W + w
  int w = row % W_;
  int tmp = row / W_;
  int i = tmp % H_;
  int b = tmp / H_;
  __shared__ float qv[C8_];
  __shared__ float ev[112];
  __shared__ float red[2];
  int t = threadIdx.x;
  if (t < C8_) qv[t] = q[((size_t)(b * C8_ + t)) * P_ + i * W_ + w];
  __syncthreads();
  if (t < 112) {
    float e;
    if (t < H_) {
      int j = t;
      if (j == i) {
        e = -1e30f;
      } else {
        float s = 0.f;
        for (int c = 0; c < C8_; ++c)
          s += qv[c] * kk_[((size_t)(b * C8_ + c)) * P_ + j * W_ + w];
        e = s;
      }
    } else {
      int kc = t - H_;
      float s = 0.f;
      for (int c = 0; c < C8_; ++c)
        s += qv[c] * kk_[((size_t)(b * C8_ + c)) * P_ + i * W_ + kc];
      e = s;
    }
    ev[t] = e;
  }
  __syncthreads();
  if (t == 0) {
    float m = -3.0e38f;
    for (int x = 0; x < 112; ++x) m = fmaxf(m, ev[x]);
    red[0] = m;
  }
  __syncthreads();
  if (t < 112) ev[t] = expf(ev[t] - red[0]);
  __syncthreads();
  if (t == 0) {
    float s = 0.f;
    for (int x = 0; x < 112; ++x) s += ev[x];
    red[1] = s;
  }
  __syncthreads();
  if (t < 112) attn[(size_t)row * 112 + t] = ev[t] / red[1];
}

// out[b,c,i,w] = sum_j v[b,c,j,w]*aH[j] + sum_k v[b,c,i,k]*aW[k]
__global__ __launch_bounds__(256) void cc_apply(
    const float* __restrict__ v, const float* __restrict__ attn,
    float* __restrict__ out) {
  int row = blockIdx.x;
  int w = row % W_;
  int tmp = row / W_;
  int i = tmp % H_;
  int b = tmp / H_;
  __shared__ float ar[112];
  int t = threadIdx.x;
  if (t < 112) ar[t] = attn[(size_t)row * 112 + t];
  __syncthreads();
  for (int c = t; c < C_; c += 256) {
    const float* __restrict__ vb = v + ((size_t)(b * C_ + c)) * P_;
    float s = 0.f;
    for (int j = 0; j < H_; ++j) s += vb[j * W_ + w] * ar[j];
    const float* __restrict__ vr = vb + i * W_;
    for (int kc = 0; kc < W_; ++kc) s += vr[kc] * ar[H_ + kc];
    out[((size_t)(b * C_ + c)) * P_ + i * W_ + w] = s;
  }
}

// ---------------------------------------------------------------------------
// S[b,c] = mean_p sum_i feats[i][b,c,p]
// ---------------------------------------------------------------------------
__global__ __launch_bounds__(256) void gap_kernel(
    const float* __restrict__ f0, const float* __restrict__ f1,
    const float* __restrict__ f2, const float* __restrict__ f3,
    float* __restrict__ S) {
  int bc = blockIdx.x;
  __shared__ float red[256];
  size_t base = (size_t)bc * P_;
  float s = 0.f;
  for (int p = threadIdx.x; p < P_; p += 256)
    s += f0[base + p] + f1[base + p] + f2[base + p] + f3[base + p];
  red[threadIdx.x] = s;
  __syncthreads();
  for (int st = 128; st > 0; st >>= 1) {
    if (threadIdx.x < st) red[threadIdx.x] += red[threadIdx.x + st];
    __syncthreads();
  }
  if (threadIdx.x == 0) S[bc] = red[0] / (float)P_;
}

// Z = S @ fc_w.T + fc_b ; weights_i = Z @ fcs_w_i.T + fcs_b_i ; softmax over i
__global__ __launch_bounds__(256) void sk_fc(
    const float* __restrict__ S, const float* __restrict__ fcw,
    const float* __restrict__ fcb,
    const float* __restrict__ fw0, const float* __restrict__ fw1,
    const float* __restrict__ fw2, const float* __restrict__ fw3,
    const float* __restrict__ fb0, const float* __restrict__ fb1,
    const float* __restrict__ fb2, const float* __restrict__ fb3,
    float* __restrict__ aw) {
  int b = blockIdx.x;
  __shared__ float Zs[D_];
  int t = threadIdx.x;
  if (t < D_) {
    float s = fcb[t];
    for (int c = 0; c < C_; ++c) s += S[b * C_ + c] * fcw[t * C_ + c];
    Zs[t] = s;
  }
  __syncthreads();
  const float* fw[4] = {fw0, fw1, fw2, fw3};
  const float* fb[4] = {fb0, fb1, fb2, fb3};
  for (int c = t; c < C_; c += 256) {
    float wv[4];
    float m = -3.0e38f;
    for (int i = 0; i < 4; ++i) {
      float s = fb[i][c];
      for (int d = 0; d < D_; ++d) s += Zs[d] * fw[i][(size_t)c * D_ + d];
      wv[i] = s;
      m = fmaxf(m, s);
    }
    float sum = 0.f;
    for (int i = 0; i < 4; ++i) { wv[i] = expf(wv[i] - m); sum += wv[i]; }
    for (int i = 0; i < 4; ++i)
      aw[((size_t)i * B_ + b) * C_ + c] = wv[i] / sum;
  }
}

__global__ __launch_bounds__(256) void sk_fuse(
    const float* __restrict__ f0, const float* __restrict__ f1,
    const float* __restrict__ f2, const float* __restrict__ f3,
    const float* __restrict__ aw, float* __restrict__ out) {
  size_t idx = (size_t)blockIdx.x * 256 + threadIdx.x;
  if (idx >= (size_t)B_ * C_ * P_) return;
  int bc = (int)(idx / P_);
  int b = bc / C_;
  int c = bc % C_;
  float s = aw[((size_t)0 * B_ + b) * C_ + c] * f0[idx]
          + aw[((size_t)1 * B_ + b) * C_ + c] * f1[idx]
          + aw[((size_t)2 * B_ + b) * C_ + c] * f2[idx]
          + aw[((size_t)3 * B_ + b) * C_ + c] * f3[idx];
  out[idx] = s;
}

// ---------------------------------------------------------------------------
// Host orchestration
// ---------------------------------------------------------------------------
static inline size_t align256(size_t x) { return (x + 255) & ~(size_t)255; }

extern "C" void kernel_launch(void* const* d_in, const int* in_sizes, int n_in,
                              void* d_out, int out_size, void* d_ws, size_t ws_size,
                              hipStream_t stream) {
  (void)in_sizes; (void)n_in; (void)out_size; (void)ws_size;
  const int KS[4] = {1, 3, 5, 7};

  const float* x   = (const float*)d_in[0];
  const float* wq  = (const float*)d_in[1];
  const float* bq  = (const float*)d_in[2];
  const float* wk  = (const float*)d_in[3];
  const float* bk  = (const float*)d_in[4];
  const float* wv  = (const float*)d_in[5];
  const float* bv  = (const float*)d_in[6];
  const float* convw[4], *convb[4], *bns[4], *bnb[4], *bnm[4], *bnv[4];
  for (int i = 0; i < 4; ++i) {
    convw[i] = (const float*)d_in[7 + i];
    convb[i] = (const float*)d_in[11 + i];
    bns[i]   = (const float*)d_in[15 + i];
    bnb[i]   = (const float*)d_in[19 + i];
    bnm[i]   = (const float*)d_in[23 + i];
    bnv[i]   = (const float*)d_in[27 + i];
  }
  const float* fcw = (const float*)d_in[31];
  const float* fcb = (const float*)d_in[32];
  const float* fw[4], *fb[4];
  for (int i = 0; i < 4; ++i) {
    fw[i] = (const float*)d_in[33 + i];
    fb[i] = (const float*)d_in[37 + i];
  }

  // ---- workspace carve ----
  char* ws = (char*)d_ws;
  size_t off = 0;
  auto carve = [&](size_t bytes) { char* p = ws + off; off += align256(bytes); return p; };

  const size_t bcp  = (size_t)B_ * C_ * P_;           // 9,633,792
  float*    ccA   = (float*)carve(bcp * 4);
  float*    ccB   = (float*)carve(bcp * 4);
  _Float16* Xt    = (_Float16*)carve(bcp * 2);
  float*    qbuf  = (float*)carve((size_t)B_ * C8_ * P_ * 4);
  float*    kbuf  = (float*)carve((size_t)B_ * C8_ * P_ * 4);
  float*    vbuf  = (float*)carve(bcp * 4);
  float*    attn  = (float*)carve((size_t)B_ * H_ * W_ * 112 * 4);
  float*    feats[4];
  for (int i = 0; i < 4; ++i) feats[i] = (float*)carve(bcp * 4);
  _Float16* wq16  = (_Float16*)carve((size_t)C8_ * C_ * 2);
  _Float16* wk16  = (_Float16*)carve((size_t)C8_ * C_ * 2);
  _Float16* wv16  = (_Float16*)carve((size_t)C_ * C_ * 2);
  _Float16* cw16[4];
  for (int i = 0; i < 4; ++i)
    cw16[i] = (_Float16*)carve((size_t)C_ * C_ * KS[i] * KS[i] * 2);
  float*    Sbuf  = (float*)carve((size_t)B_ * C_ * 4);
  float*    awbuf = (float*)carve((size_t)4 * B_ * C_ * 4);

  // ---- weight conversion (re-done each call; deterministic) ----
  {
    int n = C8_ * C_;
    cvt_f16_kernel<<<(n + 255) / 256, 256, 0, stream>>>(wq, wq16, n);
    cvt_f16_kernel<<<(n + 255) / 256, 256, 0, stream>>>(wk, wk16, n);
    int nv = C_ * C_;
    cvt_f16_kernel<<<(nv + 255) / 256, 256, 0, stream>>>(wv, wv16, nv);
    for (int i = 0; i < 4; ++i) {
      size_t np = (size_t)C_ * C_ * KS[i] * KS[i];
      pack_conv_w_kernel<<<(unsigned)((np + 255) / 256), 256, 0, stream>>>(
          convw[i], cw16[i], KS[i]);
    }
  }

  const unsigned gridXT   = (unsigned)((bcp + 255) / 256);
  const unsigned rows     = B_ * H_ * W_;              // 12544
  const unsigned gridGqk  = B_ * 1 * (P_ / 64);        // O=96, NW=6 -> MB=1
  const unsigned gridGv   = B_ * (C_ / 128) * (P_ / 64);
  const unsigned gridConv = B_ * (C_ / 64) * (P_ / 64);   // 2352
  const size_t   convLds  = (size_t)64 * C_ * sizeof(_Float16); // 96 KB

  const float* cur_in = x;
  for (int it = 0; it < 4; ++it) {
    float* cc_out = (it & 1) ? ccB : ccA;

    // 1) transpose/convert criss-cross input to f16 (B,P,C)
    cvt_xt_kernel<<<gridXT, 256, 0, stream>>>(cur_in, Xt);

    // 2) q, k, v projections (WMMA GEMM)
    gemm_bias_wmma<<<gridGqk, 192, 0, stream>>>(wq16, bq, Xt, qbuf, C8_);
    gemm_bias_wmma<<<gridGqk, 192, 0, stream>>>(wk16, bk, Xt, kbuf, C8_);
    gemm_bias_wmma<<<gridGv, 256, 0, stream>>>(wv16, bv, Xt, vbuf, C_);

    // 3) energies + softmax, 4) aggregate
    cc_energy_softmax<<<rows, 128, 0, stream>>>(qbuf, kbuf, attn);
    cc_apply<<<rows, 256, 0, stream>>>(vbuf, attn, cc_out);

    // 5) transpose cc output, 6) conv+bn+relu (WMMA implicit GEMM + TDM)
    cvt_xt_kernel<<<gridXT, 256, 0, stream>>>(cc_out, Xt);
    conv_bn_relu_wmma<<<gridConv, 128, convLds, stream>>>(
        cw16[it], convb[it], bns[it], bnb[it], bnm[it], bnv[it],
        Xt, feats[it], KS[it]);

    cur_in = cc_out;
  }

  // ---- selective-kernel fusion ----
  gap_kernel<<<B_ * C_, 256, 0, stream>>>(feats[0], feats[1], feats[2], feats[3], Sbuf);
  sk_fc<<<B_, 256, 0, stream>>>(Sbuf, fcw, fcb,
                                fw[0], fw[1], fw[2], fw[3],
                                fb[0], fb[1], fb[2], fb[3], awbuf);
  sk_fuse<<<gridXT, 256, 0, stream>>>(feats[0], feats[1], feats[2], feats[3],
                                      awbuf, (float*)d_out);
}
